// RPN_25323127177304
// MI455X (gfx1250) — compile-verified, matrix operand order checked by reading, not run on hardware
//
#include <hip/hip_runtime.h>
#include <hip/hip_bf16.h>

// ---------------------------------------------------------------------------
// RPN (Faster R-CNN region proposal network) for MI455X / gfx1250.
//   features [8,256,128,128] f32 -> 3x3 conv 256->256 + ReLU (bf16 WMMA)
//   -> 1x1 loc (36ch) & score (18ch) heads fused in-kernel (bf16 WMMA)
//   -> softmax fg, box decode/clip, radix top-12000, greedy NMS -> 2000 rois.
// Weights are staged into LDS by the Tensor Data Mover (tensor_load_to_lds,
// TENSORcnt), double-buffered against WMMA compute.
// ---------------------------------------------------------------------------

#define NIMG   8
#define CIN    256
#define HH     128
#define WW     128
#define KOUT   256
#define AANCH  9
#define HWA    (HH * WW * AANCH)        // 147456
#define NPRE   12000
#define NPOST  2000

// output element offsets (fp32 elements; roi_indices written as int32 bits)
#define LOCS_N    ((size_t)NIMG * HWA * 4)          // 4718592
#define SCORES_N  ((size_t)NIMG * HWA * 2)          // 2359296
#define ROIS_N    ((size_t)NIMG * NPOST * 4)        // 64000
#define RIDX_N    ((size_t)NIMG * NPOST)            // 16000
#define ANCH_N    ((size_t)HWA * 4)                 // 589824

// workspace byte offsets (all 16B aligned)
#define WS_WB     ((size_t)0)                        // [9][256][256] bf16  (1179648 B)
#define WS_WS     ((size_t)1179648)                  // [64][256] bf16      (32768 B)
#define WS_BIAS2  ((size_t)1212416)                  // [64] f32            (256 B)
#define WS_FG     ((size_t)1212672)                  // [8][HWA] f32
#define WS_BOXES  ((size_t)5931264)                  // [8][HWA][4] f32
#define WS_KEYS   ((size_t)24805632)                 // [8][HWA] u32
#define WS_MSC    ((size_t)29524224)                 // [8][HWA] f32 masked score
#define WS_HIST   ((size_t)34242816)                 // [8][2048] u32
#define WS_TINFO  ((size_t)34308352)                 // [8][4] i32 (tb, rem, cnt_hi, cnt_eq)
#define WS_SELB   ((size_t)34308480)                 // [8][12000][4] f32
#define WS_SELS   ((size_t)35844480)                 // [8][12000] f32

typedef __attribute__((ext_vector_type(16))) __bf16          v16bf;
typedef __attribute__((ext_vector_type(8)))  float           v8f;
typedef __attribute__((ext_vector_type(8)))  unsigned short  u16x8;
typedef __attribute__((ext_vector_type(4)))  unsigned int    u32x4;
typedef __attribute__((ext_vector_type(8)))  unsigned int    u32x8;

union BFrag { v16bf bf; u16x8 us[2]; };

__device__ __forceinline__ unsigned short f2bf(float f) {
  unsigned u = __float_as_uint(f);
  u = (u + 0x7FFFu + ((u >> 16) & 1u)) >> 16;   // round-to-nearest-even
  return (unsigned short)u;
}

// ---------------------------------------------------------------------------
// Tensor Data Mover: DMA a 2D bf16 tile [tile_d1 rows x tile_d0 elems] from a
// row-major tensor (row stride = stride0 elems) in global memory into LDS,
// packed contiguously. Descriptor per CDNA5 ISA ch.8 (D# groups 0/1, 2D form).
// Tracked with TENSORcnt.
// ---------------------------------------------------------------------------
__device__ __forceinline__ void tdm_load_2d_bf16(unsigned lds_off, const void* gptr,
                                                 unsigned tile_d0, unsigned tile_d1,
                                                 unsigned tensor_d0, unsigned tensor_d1,
                                                 unsigned stride0) {
  unsigned long long ga = (unsigned long long)(size_t)gptr;
  u32x4 g0;
  g0[0] = 1u;                                   // count=1, user descriptor
  g0[1] = lds_off;                              // lds_addr (bytes)
  g0[2] = (unsigned)ga;                         // global_addr[31:0]
  g0[3] = ((unsigned)(ga >> 32) & 0x01FFFFFFu)  // global_addr[56:32]
          | 0x80000000u;                        // type = 2 ("image")
  u32x8 g1;
  g1[0] = 0x00010000u;                          // data_size=1 (2B), no pad/multicast
  g1[1] = (tensor_d0 & 0xFFFFu) << 16;          // tensor_dim0[15:0]
  g1[2] = (tensor_d0 >> 16) | ((tensor_d1 & 0xFFFFu) << 16);   // dim0 hi | dim1 lo
  g1[3] = (tensor_d1 >> 16) | (tile_d0 << 16);  // dim1 hi | tile_dim0
  g1[4] = tile_d1 & 0xFFFFu;                    // tile_dim1 (tile_dim2 = 0)
  g1[5] = stride0;                              // tensor_dim0_stride[31:0]
  g1[6] = 0u;                                   // stride0 hi | dim1_stride lo
  g1[7] = 0u;
  asm volatile("tensor_load_to_lds %0, %1" :: "s"(g0), "s"(g1) : "memory");
}

// ---------------------------------------------------------------------------
// Weight prep: conv1 weights -> bf16 [tap][kout][c]
// ---------------------------------------------------------------------------
__global__ void prep_wB_k(const float* __restrict__ w, unsigned short* __restrict__ wB) {
  int idx = blockIdx.x * blockDim.x + threadIdx.x;
  if (idx >= 9 * KOUT * CIN) return;
  int tap = idx / (KOUT * CIN);
  int r   = idx - tap * (KOUT * CIN);
  int ko  = r >> 8;
  int c   = r & 255;
  wB[idx] = f2bf(w[((size_t)ko * CIN + c) * 9 + tap]);   // OIHW source
}

// 1x1 heads packed: kout2 [0,36)=loc, [36,54)=score, [54,64)=pad. Plus bias.
__global__ void prep_wS_k(const float* __restrict__ locw, const float* __restrict__ scorew,
                          const float* __restrict__ locb, const float* __restrict__ scoreb,
                          unsigned short* __restrict__ wS, float* __restrict__ bias2) {
  int idx = blockIdx.x * blockDim.x + threadIdx.x;
  if (idx < 64 * CIN) {
    int ko = idx >> 8, c = idx & 255;
    float v = 0.f;
    if (ko < 36)      v = locw[(size_t)ko * CIN + c];
    else if (ko < 54) v = scorew[(size_t)(ko - 36) * CIN + c];
    wS[idx] = f2bf(v);
  } else if (idx < 64 * CIN + 64) {
    int ko = idx - 64 * CIN;
    float v = 0.f;
    if (ko < 36)      v = locb[ko];
    else if (ko < 54) v = scoreb[ko - 36];
    bias2[ko] = v;
  }
}

// ---------------------------------------------------------------------------
// Fused 3x3 conv (bf16 WMMA, f32 acc) + ReLU + 1x1 loc/score heads.
// Block = 256 threads (8 waves), tile = 64 pixels (half row) x 256 channels.
// Weight tiles are DMA'd into LDS by the TDM (double-buffered per tap).
// ---------------------------------------------------------------------------
__global__ __launch_bounds__(256)
void rpn_conv_fused_k(const float* __restrict__ feat,
                      const unsigned short* __restrict__ wB,   // [9][256][256] bf16
                      const float* __restrict__ conv1_b,
                      const unsigned short* __restrict__ wS,   // [64][256] bf16
                      const float* __restrict__ bias2,         // [64]
                      float* __restrict__ out_locs,
                      float* __restrict__ out_scores) {
  __shared__ __align__(16) unsigned short in_s[3][66][40];     // [row][x][c] bf16, padded
  __shared__ __align__(16) unsigned short mid_s[64][264];      // [px][kout] bf16, padded
  __shared__ __align__(16) unsigned short w_s[2][KOUT * 32];   // TDM weight tiles [kout][c]
  __shared__ __align__(16) unsigned short hw_s[64 * CIN];      // TDM 1x1-head weights [ko][c]

  const int tid = threadIdx.x;
  const int w   = tid >> 5;        // wave 0..7 -> 32-channel slice
  const int l   = tid & 31;
  const int lh  = l >> 4;          // K-half select per WMMA layout
  const int ll  = l & 15;
  const int bc  = lh * 8;          // base K element within 32-slice

  const int b  = blockIdx.x;
  const int n  = b >> 8;
  const int y  = (b & 255) >> 1;
  const int x0 = (b & 1) * 64;

  // Kick off the 1x1-head weight slab DMA once (64x256 bf16 = 32 KB).
  if (w == 0) {
    tdm_load_2d_bf16((unsigned)(size_t)&hw_s[0], wS,
                     /*tile*/ CIN, 64, /*tensor*/ CIN, 64, /*stride0*/ CIN);
  }

  const v8f vzero = {};
  v8f acc[4][2];
#pragma unroll
  for (int m = 0; m < 4; ++m)
#pragma unroll
    for (int s = 0; s < 2; ++s) acc[m][s] = vzero;

  for (int cb = 0; cb < 8; ++cb) {                 // 8 x 32-channel K steps
    // ---- TDM: first weight tile of this c-step (tap 0) into buffer 0 ----
    if (w == 0) {
      tdm_load_2d_bf16((unsigned)(size_t)&w_s[0][0],
                       wB + ((size_t)0 * KOUT * CIN + (size_t)cb * 32),
                       /*tile*/ 32, KOUT, /*tensor*/ CIN, 9 * KOUT, /*stride0*/ CIN);
    }

    // ---- stage 32 channels x 3 rows x 66 px into LDS as bf16 (VALU cvt) ----
    for (int idx = tid; idx < 3 * 66 * 32; idx += 256) {
      int r  = idx / (66 * 32);
      int t  = idx - r * (66 * 32);
      int ci = t / 66;
      int xi = t - ci * 66;
      int gy = y + r - 1;
      int gx = x0 + xi - 1;
      float v = 0.f;
      if ((unsigned)gy < HH && (unsigned)gx < WW)
        v = feat[(((size_t)n * CIN + (size_t)(cb * 32 + ci)) * HH + gy) * WW + gx];
      in_s[r][xi][ci] = f2bf(v);
    }
    if (cb < 7)  // global_prefetch_b8 of the next channel slice
      __builtin_prefetch(&feat[(((size_t)n * CIN + (size_t)((cb + 1) * 32)) * HH + y) * WW + x0], 0, 1);

#pragma unroll
    for (int tap = 0; tap < 9; ++tap) {
      const int dy  = tap / 3;
      const int dx  = tap - 3 * dy;
      const int buf = tap & 1;
      // Current tap's weight tile must be resident before anyone reads it.
      if (w == 0) __builtin_amdgcn_s_wait_tensorcnt((short)0);
      __syncthreads();
      // Overlap: DMA the next tap's tile into the other buffer.
      if (w == 0 && tap < 8) {
        tdm_load_2d_bf16((unsigned)(size_t)&w_s[buf ^ 1][0],
                         wB + ((size_t)(tap + 1) * KOUT * CIN + (size_t)cb * 32),
                         /*tile*/ 32, KOUT, /*tensor*/ CIN, 9 * KOUT, /*stride0*/ CIN);
      }

      BFrag bB[2];
#pragma unroll
      for (int s = 0; s < 2; ++s) {                // 2 x 16 output channels
        const int kout = w * 32 + s * 16 + ll;
        const unsigned short* wp = &w_s[buf][kout * 32];
        bB[s].us[0] = *(const u16x8*)(wp + bc);
        bB[s].us[1] = *(const u16x8*)(wp + 16 + bc);
      }
#pragma unroll
      for (int m = 0; m < 4; ++m) {                // 4 x 16 pixels
        BFrag aA;
        const int xi = m * 16 + ll + dx;
        aA.us[0] = *(const u16x8*)&in_s[dy][xi][bc];
        aA.us[1] = *(const u16x8*)&in_s[dy][xi][16 + bc];
#pragma unroll
        for (int s = 0; s < 2; ++s)
          acc[m][s] = __builtin_amdgcn_wmma_f32_16x16x32_bf16(
              false, aA.bf, false, bB[s].bf, (short)0, acc[m][s], false, false);
      }
    }
    __syncthreads();   // protect in_s before restaging
  }

  // ---- bias + ReLU -> LDS mid tile (bf16) ----
#pragma unroll
  for (int s = 0; s < 2; ++s) {
    const int kout = w * 32 + s * 16 + ll;
    const float bias = conv1_b[kout];
#pragma unroll
    for (int m = 0; m < 4; ++m)
#pragma unroll
      for (int v = 0; v < 8; ++v) {
        const int px = m * 16 + v + 8 * lh;
        float val = acc[m][s][v] + bias;
        mid_s[px][kout] = f2bf(fmaxf(val, 0.f));
      }
  }
  __syncthreads();

  // ---- fused 1x1 heads: 64 output channels (36 loc + 18 score + pad) ----
  const int m2 = w & 3;       // pixel sub-tile
  const int sp = w >> 2;      // which 32-channel half of the head outputs
  v8f acc2[2];
  acc2[0] = vzero; acc2[1] = vzero;
#pragma unroll
  for (int kb = 0; kb < 8; ++kb) {
    BFrag aA;
    const int px = m2 * 16 + ll;
    aA.us[0] = *(const u16x8*)&mid_s[px][kb * 32 + bc];
    aA.us[1] = *(const u16x8*)&mid_s[px][kb * 32 + 16 + bc];
#pragma unroll
    for (int s2 = 0; s2 < 2; ++s2) {
      const int ko = (sp * 2 + s2) * 16 + ll;
      const unsigned short* wp = &hw_s[ko * CIN + kb * 32];
      BFrag bB;
      bB.us[0] = *(const u16x8*)(wp + bc);
      bB.us[1] = *(const u16x8*)(wp + 16 + bc);
      acc2[s2] = __builtin_amdgcn_wmma_f32_16x16x32_bf16(
          false, aA.bf, false, bB.bf, (short)0, acc2[s2], false, false);
    }
  }

  // ---- scatter head outputs: locs [N,HWA,4], scores [N,HWA,2] ----
#pragma unroll
  for (int s2 = 0; s2 < 2; ++s2) {
    const int ko = (sp * 2 + s2) * 16 + ll;
    const float bb = bias2[ko];
#pragma unroll
    for (int v = 0; v < 8; ++v) {
      const int px  = m2 * 16 + v + 8 * lh;
      const int pix = y * WW + x0 + px;
      const float val = acc2[s2][v] + bb;
      if (ko < 36) {
        const int ai = ko >> 2, j = ko & 3;
        out_locs[((size_t)n * HWA + (size_t)pix * AANCH + ai) * 4 + j] = val;
      } else if (ko < 54) {
        const int c2 = ko - 36, ai = c2 >> 1, j = c2 & 1;
        out_scores[((size_t)n * HWA + (size_t)pix * AANCH + ai) * 2 + j] = val;
      }
    }
  }
}

// ---------------------------------------------------------------------------
// Anchors [HWA,4]
// ---------------------------------------------------------------------------
__global__ void anchors_k(float* __restrict__ anch) {
  int i = blockIdx.x * blockDim.x + threadIdx.x;
  if (i >= HWA) return;
  int cell = i / AANCH, a = i - AANCH * cell;
  int ri = a / 3, si = a - 3 * ri;
  const float ratios[3] = {0.5f, 1.0f, 2.0f};
  const float scales[3] = {8.0f, 16.0f, 32.0f};
  float r = ratios[ri], s = scales[si];
  float h  = 16.f * s * sqrtf(r);
  float wd = 16.f * s * sqrtf(1.f / r);
  float sy = (float)((cell / WW) * 16);
  float sx = (float)((cell % WW) * 16);
  anch[4 * i + 0] = 8.f - 0.5f * h  + sy;
  anch[4 * i + 1] = 8.f - 0.5f * wd + sx;
  anch[4 * i + 2] = 8.f + 0.5f * h  + sy;
  anch[4 * i + 3] = 8.f + 0.5f * wd + sx;
}

// fg = softmax(scores)[...,1]
__global__ void fg_softmax_k(const float* __restrict__ scores, float* __restrict__ fg, int total) {
  int i = blockIdx.x * blockDim.x + threadIdx.x;
  if (i >= total) return;
  float s0 = scores[2 * i], s1 = scores[2 * i + 1];
  fg[i] = 1.f / (1.f + __expf(s0 - s1));
}

// decode + clip + min-size mask; emit boxes, orderable keys, masked scores
__global__ void decode_k(const float* __restrict__ locs, const float* __restrict__ fg,
                         const float* __restrict__ anch,
                         const int* __restrict__ ph, const int* __restrict__ pw,
                         const int* __restrict__ psc,
                         float* __restrict__ boxes, unsigned* __restrict__ keys,
                         float* __restrict__ msc, int total) {
  int idx = blockIdx.x * blockDim.x + threadIdx.x;
  if (idx >= total) return;
  int i = idx % HWA;
  float a0 = anch[4 * i], a1 = anch[4 * i + 1], a2 = anch[4 * i + 2], a3 = anch[4 * i + 3];
  const float* L = locs + (size_t)idx * 4;
  float h = a2 - a0, wd = a3 - a1;
  float cy = a0 + 0.5f * h, cx = a1 + 0.5f * wd;
  float ncy = L[0] * h + cy, ncx = L[1] * wd + cx;
  float nh = __expf(L[2]) * h, nw = __expf(L[3]) * wd;
  float imh = (float)ph[0], imw = (float)pw[0];
  float y1 = fminf(fmaxf(ncy - 0.5f * nh, 0.f), imh);
  float x1 = fminf(fmaxf(ncx - 0.5f * nw, 0.f), imw);
  float y2 = fminf(fmaxf(ncy + 0.5f * nh, 0.f), imh);
  float x2 = fminf(fmaxf(ncx + 0.5f * nw, 0.f), imw);
  float mins = 16.f * (float)psc[0];
  bool ok = ((y2 - y1) >= mins) && ((x2 - x1) >= mins);
  float sc = ok ? fg[idx] : -INFINITY;
  size_t o = (size_t)idx * 4;
  boxes[o + 0] = y1; boxes[o + 1] = x1; boxes[o + 2] = y2; boxes[o + 3] = x2;
  msc[idx] = sc;
  unsigned u = __float_as_uint(sc);
  keys[idx] = u ^ (((int)u >> 31) ? 0xFFFFFFFFu : 0x80000000u);
}

__global__ void hist_k(const unsigned* __restrict__ keys, unsigned* __restrict__ hist, int total) {
  int idx = blockIdx.x * blockDim.x + threadIdx.x;
  if (idx >= total) return;
  int n = idx / HWA;
  atomicAdd(&hist[(size_t)n * 2048 + (keys[idx] >> 21)], 1u);
}

__global__ void thresh_k(const unsigned* __restrict__ hist, int* __restrict__ tinfo) {
  int n = blockIdx.x;
  const unsigned* h = hist + (size_t)n * 2048;
  unsigned cum = 0; int b;
  for (b = 2047; b >= 0; --b) { cum += h[b]; if (cum >= NPRE) break; }
  if (b < 0) b = 0;
  unsigned above = cum - h[b];
  tinfo[n * 4 + 0] = b;
  tinfo[n * 4 + 1] = NPRE - (int)above;   // remaining slots in threshold bin
}

__global__ void initsel_k(float* __restrict__ selb, float* __restrict__ sels) {
  int idx = blockIdx.x * blockDim.x + threadIdx.x;
  if (idx >= NIMG * NPRE) return;
  sels[idx] = -INFINITY;
  size_t o = (size_t)idx * 4;
  selb[o + 0] = 0.f; selb[o + 1] = 0.f; selb[o + 2] = 0.f; selb[o + 3] = 0.f;
}

__global__ void compact_k(const unsigned* __restrict__ keys, const float* __restrict__ msc,
                          const float* __restrict__ boxes, int* __restrict__ tinfo,
                          float* __restrict__ selb, float* __restrict__ sels, int total) {
  int idx = blockIdx.x * blockDim.x + threadIdx.x;
  if (idx >= total) return;
  int n = idx / HWA;
  unsigned key = keys[idx];
  int bin = (int)(key >> 21);
  int tb  = tinfo[n * 4 + 0];
  int rem = tinfo[n * 4 + 1];
  int pos = -1;
  if (bin > tb) {
    pos = atomicAdd(&tinfo[n * 4 + 2], 1);
  } else if (bin == tb) {
    int s = atomicAdd(&tinfo[n * 4 + 3], 1);
    if (s < rem) pos = (NPRE - rem) + s;
  }
  if (pos >= 0 && pos < NPRE) {
    size_t o = (size_t)n * NPRE + pos;
    sels[o] = msc[idx];
    size_t bsrc = (size_t)idx * 4, bdst = o * 4;
    selb[bdst + 0] = boxes[bsrc + 0];
    selb[bdst + 1] = boxes[bsrc + 1];
    selb[bdst + 2] = boxes[bsrc + 2];
    selb[bdst + 3] = boxes[bsrc + 3];
  }
}

// ---------------------------------------------------------------------------
// Greedy NMS: one block per image, 2000 rounds of argmax + IoU suppression.
// ---------------------------------------------------------------------------
__global__ __launch_bounds__(256)
void nms_k(const float* __restrict__ selb, float* __restrict__ sels,
           float* __restrict__ rois, int* __restrict__ ridx) {
  const int n = blockIdx.x, tid = threadIdx.x;
  const float* B = selb + (size_t)n * NPRE * 4;
  float* S = sels + (size_t)n * NPRE;
  __shared__ float sv[256];
  __shared__ int   si[256];
  __shared__ float sbox[4];
  __shared__ int   sval;
  for (int it = 0; it < NPOST; ++it) {
    float best = -INFINITY; int bi = -1;
    for (int i = tid; i < NPRE; i += 256) {
      float s = S[i];
      if (s > best) { best = s; bi = i; }
    }
    sv[tid] = best; si[tid] = bi;
    __syncthreads();
    for (int off = 128; off > 0; off >>= 1) {
      if (tid < off && sv[tid + off] > sv[tid]) { sv[tid] = sv[tid + off]; si[tid] = si[tid + off]; }
      __syncthreads();
    }
    if (tid == 0) {
      int j = si[0];
      int valid = (j >= 0) && (sv[0] > -1e37f);
      sval = valid;
      size_t ro = ((size_t)n * NPOST + it) * 4;
      if (valid) {
        sbox[0] = B[4 * (size_t)j + 0]; sbox[1] = B[4 * (size_t)j + 1];
        sbox[2] = B[4 * (size_t)j + 2]; sbox[3] = B[4 * (size_t)j + 3];
        rois[ro + 0] = sbox[0]; rois[ro + 1] = sbox[1];
        rois[ro + 2] = sbox[2]; rois[ro + 3] = sbox[3];
      } else {
        sbox[0] = sbox[1] = sbox[2] = sbox[3] = 0.f;
        rois[ro + 0] = 0.f; rois[ro + 1] = 0.f; rois[ro + 2] = 0.f; rois[ro + 3] = 0.f;
      }
      ridx[(size_t)n * NPOST + it] = n;
    }
    __syncthreads();
    if (sval) {
      float by1 = sbox[0], bx1 = sbox[1], by2 = sbox[2], bx2 = sbox[3];
      float ba = (by2 - by1) * (bx2 - bx1);
      for (int i = tid; i < NPRE; i += 256) {
        float s = S[i];
        if (s <= -1e37f) continue;
        float iy1 = fmaxf(by1, B[4 * (size_t)i + 0]);
        float ix1 = fmaxf(bx1, B[4 * (size_t)i + 1]);
        float iy2 = fminf(by2, B[4 * (size_t)i + 2]);
        float ix2 = fminf(bx2, B[4 * (size_t)i + 3]);
        float inter = fmaxf(iy2 - iy1, 0.f) * fmaxf(ix2 - ix1, 0.f);
        float a2 = (B[4 * (size_t)i + 2] - B[4 * (size_t)i + 0]) *
                   (B[4 * (size_t)i + 3] - B[4 * (size_t)i + 1]);
        float iou = inter / (ba + a2 - inter + 1e-9f);
        if (iou > 0.7f) S[i] = -INFINITY;
      }
    }
    __syncthreads();
  }
}

// ---------------------------------------------------------------------------
extern "C" void kernel_launch(void* const* d_in, const int* in_sizes, int n_in,
                              void* d_out, int out_size, void* d_ws, size_t ws_size,
                              hipStream_t stream) {
  (void)in_sizes; (void)n_in; (void)out_size; (void)ws_size;

  const float* feat   = (const float*)d_in[0];
  const float* conv1w = (const float*)d_in[1];
  const float* conv1b = (const float*)d_in[2];
  const float* scorew = (const float*)d_in[3];
  const float* scoreb = (const float*)d_in[4];
  const float* locw   = (const float*)d_in[5];
  const float* locb   = (const float*)d_in[6];
  const int*   ph     = (const int*)d_in[7];
  const int*   pw     = (const int*)d_in[8];
  const int*   psc    = (const int*)d_in[9];

  float* out        = (float*)d_out;
  float* out_locs   = out;
  float* out_scores = out + LOCS_N;
  float* out_rois   = out + LOCS_N + SCORES_N;
  int*   out_ridx   = (int*)(out + LOCS_N + SCORES_N + ROIS_N);
  float* out_anch   = out + LOCS_N + SCORES_N + ROIS_N + RIDX_N;

  char* ws = (char*)d_ws;
  unsigned short* wB    = (unsigned short*)(ws + WS_WB);
  unsigned short* wS    = (unsigned short*)(ws + WS_WS);
  float*          bias2 = (float*)(ws + WS_BIAS2);
  float*          fg    = (float*)(ws + WS_FG);
  float*          boxes = (float*)(ws + WS_BOXES);
  unsigned*       keys  = (unsigned*)(ws + WS_KEYS);
  float*          msc   = (float*)(ws + WS_MSC);
  unsigned*       hist  = (unsigned*)(ws + WS_HIST);
  int*            tinfo = (int*)(ws + WS_TINFO);
  float*          selb  = (float*)(ws + WS_SELB);
  float*          sels  = (float*)(ws + WS_SELS);

  const int total = NIMG * HWA;   // 1179648

  // 1) weight prep (bf16 pack)
  prep_wB_k<<<(9 * KOUT * CIN + 255) / 256, 256, 0, stream>>>(conv1w, wB);
  prep_wS_k<<<(64 * CIN + 64 + 255) / 256, 256, 0, stream>>>(locw, scorew, locb, scoreb, wS, bias2);

  // 2) fused 3x3 conv + ReLU + 1x1 heads (WMMA + TDM)
  rpn_conv_fused_k<<<NIMG * HH * 2, 256, 0, stream>>>(feat, wB, conv1b, wS, bias2,
                                                      out_locs, out_scores);

  // 3) anchors + softmax fg + decode
  anchors_k<<<(HWA + 255) / 256, 256, 0, stream>>>(out_anch);
  fg_softmax_k<<<(total + 255) / 256, 256, 0, stream>>>(out_scores, fg, total);
  decode_k<<<(total + 255) / 256, 256, 0, stream>>>(out_locs, fg, out_anch, ph, pw, psc,
                                                    boxes, keys, msc, total);

  // 4) radix top-12000 per image
  hipMemsetAsync(ws + WS_HIST, 0, 8 * 2048 * sizeof(unsigned) + 8 * 4 * sizeof(int), stream);
  hist_k<<<(total + 255) / 256, 256, 0, stream>>>(keys, hist, total);
  thresh_k<<<NIMG, 1, 0, stream>>>(hist, tinfo);
  initsel_k<<<(NIMG * NPRE + 255) / 256, 256, 0, stream>>>(selb, sels);
  compact_k<<<(total + 255) / 256, 256, 0, stream>>>(keys, msc, boxes, tinfo, selb, sels, total);

  // 5) greedy NMS -> rois / roi_indices
  nms_k<<<NIMG, 256, 0, stream>>>(selb, sels, out_rois, out_ridx);
}
